// DivFeatConv_12790412607512
// MI455X (gfx1250) — compile-verified, matrix operand order checked by reading, not run on hardware
//
#include <hip/hip_runtime.h>
#include <hip/hip_bf16.h>

typedef __attribute__((ext_vector_type(16))) __bf16 v16bf;
typedef __attribute__((ext_vector_type(8)))  float  v8f;

#define N_NODES 40000
#define N_EDGES 640000
#define D 128

// ---------------------------------------------------------------------------
// Kernel 1: scatter-mean accumulation. One wave (32 lanes) per edge.
// Each lane gathers a float4 (4 contiguous feats) and does 4 f32 atomic adds.
// feat is 20.5 MB -> resident in the 192MB L2 after first touch; atomics
// resolve at L2. Lane 0 bumps the degree counter.
// ---------------------------------------------------------------------------
__global__ void sage_scatter_kernel(const float* __restrict__ feat,
                                    const int* __restrict__ src,
                                    const int* __restrict__ dst,
                                    float* __restrict__ msg,
                                    float* __restrict__ deg,
                                    int n_edges) {
    int gwave = (blockIdx.x * blockDim.x + threadIdx.x) >> 5;
    int lane  = threadIdx.x & 31;
    if (gwave >= n_edges) return;

    int s = src[gwave];
    int d = dst[gwave];

    const float4* fs = reinterpret_cast<const float4*>(feat + (size_t)s * D);
    float4 v = fs[lane];

    float* m = msg + (size_t)d * D + lane * 4;
    atomicAdd(m + 0, v.x);
    atomicAdd(m + 1, v.y);
    atomicAdd(m + 2, v.z);
    atomicAdd(m + 3, v.w);
    if (lane == 0) atomicAdd(deg + d, 1.0f);
}

// ---------------------------------------------------------------------------
// Kernel 2: fused dual GEMM + bias + relu via v_wmma_f32_16x16x32_bf16.
// Block = 256 threads = 8 waves. Block b owns node tile tm=b (16 rows);
// wave w owns output-channel tile tn=w (16 cols). K=128 in four 32-chunks,
// two WMMAs per chunk (self path + mean-neighbor path), f32 accumulation.
// ---------------------------------------------------------------------------
__global__ void __launch_bounds__(256)
sage_wmma_kernel(const float* __restrict__ feat,
                 const float* __restrict__ msg,
                 const float* __restrict__ deg,
                 const float* __restrict__ Wself,
                 const float* __restrict__ bself,
                 const float* __restrict__ Wneigh,
                 const float* __restrict__ bneigh,
                 float* __restrict__ out,
                 int n_node_tiles) {
    int tm   = blockIdx.x;            // node tile (16 rows)
    int tn   = threadIdx.x >> 5;      // out-channel tile (16 cols), wave id
    int lane = threadIdx.x & 31;
    if (tm >= n_node_tiles) return;   // wave-uniform -> EXEC all-1s at WMMA

    int half = lane >> 4;             // 0: K-low half of pairs, 1: K-high
    int r    = lane & 15;

    int arow = tm * 16 + r;           // node row this lane feeds into A
    int brow = tn * 16 + r;           // W row (= output channel) for B

    const float* fA = feat   + (size_t)arow * D;
    const float* mA = msg    + (size_t)arow * D;
    const float* wS = Wself  + (size_t)brow * D;
    const float* wN = Wneigh + (size_t)brow * D;

    float invdeg = 1.0f / fmaxf(deg[arow], 1.0f);   // zero-in-degree -> msg==0

    v8f c = {};
#pragma unroll
    for (int kb = 0; kb < D; kb += 32) {
        v16bf a_s, a_n, b_s, b_n;
#pragma unroll
        for (int vg = 0; vg < 8; ++vg) {
            // 16-bit A-fragment layout: VGPR vg holds K pair at
            // grp*16 + half*8 + wi*2 (grp = vg/4, wi = vg%4)
            int grp = vg >> 2;
            int wi  = vg & 3;
            int k   = kb + grp * 16 + half * 8 + wi * 2;

            float2 fs = *reinterpret_cast<const float2*>(fA + k);
            float2 fm = *reinterpret_cast<const float2*>(mA + k);
            float2 ws = *reinterpret_cast<const float2*>(wS + k);
            float2 wn = *reinterpret_cast<const float2*>(wN + k);

            a_s[2 * vg]     = (__bf16)fs.x;
            a_s[2 * vg + 1] = (__bf16)fs.y;
            a_n[2 * vg]     = (__bf16)(fm.x * invdeg);
            a_n[2 * vg + 1] = (__bf16)(fm.y * invdeg);
            b_s[2 * vg]     = (__bf16)ws.x;
            b_s[2 * vg + 1] = (__bf16)ws.y;
            b_n[2 * vg]     = (__bf16)wn.x;
            b_n[2 * vg + 1] = (__bf16)wn.y;
        }
        // (neg_a, A, neg_b, B, c_mod, C, reuse_a, reuse_b)
        c = __builtin_amdgcn_wmma_f32_16x16x32_bf16(false, a_s, false, b_s,
                                                    (short)0, c, false, false);
        c = __builtin_amdgcn_wmma_f32_16x16x32_bf16(false, a_n, false, b_n,
                                                    (short)0, c, false, false);
    }

    int n = tn * 16 + r;
    float bias = bself[n] + bneigh[n];
#pragma unroll
    for (int v = 0; v < 8; ++v) {
        int m = tm * 16 + v + 8 * half;   // C/D layout: VGPR v -> row v + 8*half
        out[(size_t)m * D + n] = fmaxf(c[v] + bias, 0.0f);
    }
}

extern "C" void kernel_launch(void* const* d_in, const int* in_sizes, int n_in,
                              void* d_out, int out_size, void* d_ws, size_t ws_size,
                              hipStream_t stream) {
    const float* feat   = (const float*)d_in[0];
    const int*   src    = (const int*)  d_in[1];
    const int*   dst    = (const int*)  d_in[2];
    const float* Wself  = (const float*)d_in[3];
    const float* bself  = (const float*)d_in[4];
    const float* Wneigh = (const float*)d_in[5];
    const float* bneigh = (const float*)d_in[6];
    float*       out    = (float*)d_out;

    const int n_nodes = in_sizes[0] / D;   // 40000
    const int n_edges = in_sizes[1];       // 640000

    // Workspace: msg[N_NODES*D] f32, then deg[N_NODES] f32.
    float* msg = (float*)d_ws;
    float* deg = msg + (size_t)n_nodes * D;
    size_t ws_bytes = ((size_t)n_nodes * D + n_nodes) * sizeof(float);
    hipMemsetAsync(d_ws, 0, ws_bytes, stream);

    // Scatter: 8 edges per 256-thread block (1 wave per edge).
    int edge_blocks = (n_edges + 7) / 8;
    sage_scatter_kernel<<<edge_blocks, 256, 0, stream>>>(feat, src, dst, msg, deg, n_edges);

    // GEMM: one block per 16-row node tile; 8 waves cover the 8 col-tiles.
    int n_node_tiles = n_nodes / 16;       // 2500
    sage_wmma_kernel<<<n_node_tiles, 256, 0, stream>>>(
        feat, msg, deg, Wself, bself, Wneigh, bneigh, out, n_node_tiles);
}
